// MiniMaxLightningAttention_22617297780931
// MI455X (gfx1250) — compile-verified, matrix operand order checked by reading, not run on hardware
//
#include <hip/hip_runtime.h>
#include <hip/hip_bf16.h>
#include <stdint.h>

// ---------------- problem constants (match reference) ----------------
#define BATCH    2
#define SEQLEN   4096
#define HIDDEN   4096
#define HEADS    32
#define HEAD_DIM 128
#define BLOCKSZ  256
#define NBLK     (SEQLEN / BLOCKSZ)      // 16
#define QKV_LD   (3 * HEADS * HEAD_DIM)  // 12288
#define TOKENS   (BATCH * SEQLEN)        // 8192

typedef __attribute__((ext_vector_type(16))) __bf16 v16bf;
typedef __attribute__((ext_vector_type(8)))  float  v8f;

union Frag16 {
  v16bf  v;
  uint4  q[2];
  __bf16 b[16];
};

// A-fragment (16x32 bf16) per CDNA5 ISA layout:
//   lane: m = lane%16, half = lane/16
//   elements 0..7  = K = half*8 + 0..7
//   elements 8..15 = K = 16 + half*8 + 0..7
// B-fragment (32x16) is symmetric: lane holds column n = lane%16 of B with the
// same K pattern -> loading row n of W (B = W^T) uses the identical loader.
__device__ __forceinline__ v16bf frag_row_ld(const __bf16* row, int colbase, int half) {
  Frag16 f;
  f.q[0] = *(const uint4*)(row + colbase + half * 8);
  f.q[1] = *(const uint4*)(row + colbase + 16 + half * 8);
  return f.v;
}

__device__ __forceinline__ v8f wmma_bf16(v16bf a, v16bf b, v8f c) {
  // (neg_a, A, neg_b, B, c_mod, C, reuse_a, reuse_b)
  return __builtin_amdgcn_wmma_f32_16x16x32_bf16(false, a, false, b, (short)0, c, false, false);
}

// ---- CDNA5 async global->LDS copy (ASYNCcnt-tracked DMA, bypasses VGPRs) ----
// LDS destination address comes from a VGPR; low 32 bits of a generic shared pointer are
// the wave-relative LDS address (ISA 10.2 aperture rules).
__device__ __forceinline__ void async_g2lds_b128(void* lds_dst, const void* gsrc) {
  uint32_t l = (uint32_t)(uintptr_t)lds_dst;
  asm volatile("global_load_async_to_lds_b128 %0, %1, off"
               :: "v"(l), "v"(gsrc) : "memory");
}
__device__ __forceinline__ void wait_async0() {
  asm volatile("s_wait_asynccnt 0x0" ::: "memory");
}

// ---------------- fp32 -> bf16 conversion ----------------
__global__ void f32_to_bf16_k(const float* __restrict__ in, __bf16* __restrict__ out, size_t n) {
  size_t i = (size_t)blockIdx.x * blockDim.x + threadIdx.x;
  size_t stride = (size_t)gridDim.x * blockDim.x;
  for (; i < n; i += stride) out[i] = (__bf16)in[i];
}

// ---------------- generic bf16 WMMA GEMM:  C = act(A[M,K] @ W[N,K]^T) -------
// Block tile 128x128, 256 threads = 8 waves (2x4), wave tile 64x32 (4x2 WMMA
// tiles). LDS ping-pong: async-DMA chunk k+1 into buf[nxt] while WMMAs consume
// buf[cur]; one s_wait_asynccnt + barrier per 32-wide K chunk.
// ACT: 0 = SiLU -> bf16, 1 = sigmoid -> bf16, 2 = none -> fp32.
template <int ACT, typename OutT>
__global__ __launch_bounds__(256) void gemm_bf16_k(const __bf16* __restrict__ A,
                                                   const __bf16* __restrict__ W,
                                                   OutT* __restrict__ C,
                                                   int M, int N, int K) {
  __shared__ __bf16 As[2][128][40];   // +8 pad: 16 frag rows hit distinct banks
  __shared__ __bf16 Bs[2][128][40];
  const int nb = blockIdx.x, mb = blockIdx.y;
  const int tid = threadIdx.x;
  const int w = tid >> 5, lane = tid & 31;
  const int wm = w >> 2, wn = w & 3;
  const int half = lane >> 4, ln = lane & 15;

  v8f zero8 = {0.f, 0.f, 0.f, 0.f, 0.f, 0.f, 0.f, 0.f};
  v8f acc[4][2];
#pragma unroll
  for (int i = 0; i < 4; ++i)
#pragma unroll
    for (int j = 0; j < 2; ++j) acc[i][j] = zero8;

  const int lr = tid >> 1, lc = (tid & 1) * 16;
  const __bf16* Ap = A + (size_t)(mb * 128 + lr) * K + lc;
  const __bf16* Wp = W + (size_t)(nb * 128 + lr) * K + lc;

  // prime buffer 0 via async DMA
  async_g2lds_b128(&As[0][lr][lc],     Ap);
  async_g2lds_b128(&As[0][lr][lc + 8], Ap + 8);
  async_g2lds_b128(&Bs[0][lr][lc],     Wp);
  async_g2lds_b128(&Bs[0][lr][lc + 8], Wp + 8);
  wait_async0();
  __syncthreads();

  const int nchunks = K >> 5;
  for (int ch = 0; ch < nchunks; ++ch) {
    const int cur = ch & 1, nxt = cur ^ 1;
    const bool has_next = (ch + 1) < nchunks;
    if (has_next) {                      // DMA next chunk while computing
      const int kc = (ch + 1) << 5;
      async_g2lds_b128(&As[nxt][lr][lc],     Ap + kc);
      async_g2lds_b128(&As[nxt][lr][lc + 8], Ap + kc + 8);
      async_g2lds_b128(&Bs[nxt][lr][lc],     Wp + kc);
      async_g2lds_b128(&Bs[nxt][lr][lc + 8], Wp + kc + 8);
      __builtin_prefetch(Ap + kc + 32, 0, 1);   // -> global_prefetch_b8
      __builtin_prefetch(Wp + kc + 32, 0, 1);
    }
    v16bf af[4], bfr[2];
#pragma unroll
    for (int mt = 0; mt < 4; ++mt)
      af[mt] = frag_row_ld(&As[cur][wm * 64 + mt * 16 + ln][0], 0, half);
#pragma unroll
    for (int nt = 0; nt < 2; ++nt)
      bfr[nt] = frag_row_ld(&Bs[cur][wn * 32 + nt * 16 + ln][0], 0, half);
#pragma unroll
    for (int mt = 0; mt < 4; ++mt)
#pragma unroll
      for (int nt = 0; nt < 2; ++nt) acc[mt][nt] = wmma_bf16(af[mt], bfr[nt], acc[mt][nt]);
    if (has_next) {
      wait_async0();
      __syncthreads();
    }
  }

#pragma unroll
  for (int mt = 0; mt < 4; ++mt)
#pragma unroll
    for (int nt = 0; nt < 2; ++nt)
#pragma unroll
      for (int r = 0; r < 8; ++r) {
        int row = mb * 128 + wm * 64 + mt * 16 + half * 8 + r;
        int col = nb * 128 + wn * 32 + nt * 16 + ln;
        float x = acc[mt][nt][r];
        float y;
        if (ACT == 0)       y = x / (1.f + __expf(-x));      // SiLU
        else if (ACT == 1)  y = 1.f / (1.f + __expf(-x));    // sigmoid
        else                y = x;
        C[(size_t)row * N + col] = (OutT)y;
      }
}

// ---------------- lightning attention scan ----------------
// One workgroup per (batch, head): 512 threads = 16 waves, wave w owns query
// rows [w*16, w*16+16) of each 256-token block. kv state (stored transposed,
// rows e) lives in LDS fp32 (+ bf16 mirror for WMMA B-fragments) across the
// 16-block sequential scan. All decay factors except block_decay come from one LDS table
// qexp[i] = exp(-slope*i), since slope is constant per workgroup.
__global__ __launch_bounds__(512) void lightning_attn_k(const __bf16* __restrict__ qkv,
                                                        float* __restrict__ attn_out) {
  __shared__ __bf16 vT_s[128][264];     // v^T: [e][m]
  __shared__ __bf16 kdT_s[128][264];    // (k * k_decay)^T: [d][m]
  __shared__ float  kvT_f[128][132];    // kv^T fp32 state: [e][d]
  __shared__ __bf16 kvT_b[128][136];    // bf16 mirror of kv^T for B-fragments
  __shared__ __bf16 sc_s[16][16][40];   // per-wave decayed-score scratch (16x32)
  __shared__ float  qexp_s[260];        // qexp[i] = exp(-slope*i), i in [0,256]

  const int bh = blockIdx.x;
  const int b = bh >> 5, h = bh & 31;
  const int tid = threadIdx.x;
  const int w = tid >> 5, lane = tid & 31;
  const int half = lane >> 4, ln = lane & 15;

  const float factor = 1.0f - 1.0f / (31.0f + 1e-5f) + 1e-5f;   // LAYER_IDX=1, N_LAYERS=32
  const float slope = exp2f(-0.25f * (float)(h + 1)) * factor;  // base^(h+1)*factor
  const float block_decay = __expf(-slope * (float)BLOCKSZ);
  const int ld = QKV_LD;
  const __bf16* base = qkv + (size_t)b * SEQLEN * ld + (size_t)h * 3 * HEAD_DIM;

  for (int i = tid; i < 257; i += 512) qexp_s[i] = __expf(-slope * (float)i);
  for (int i = tid; i < 128 * 132; i += 512) ((float*)kvT_f)[i] = 0.f;
  for (int i = tid; i < 128 * 136; i += 512) ((__bf16*)kvT_b)[i] = (__bf16)0.f;
  __syncthreads();

  v8f zero8 = {0.f, 0.f, 0.f, 0.f, 0.f, 0.f, 0.f, 0.f};

  for (int t = 0; t < NBLK; ++t) {
    const __bf16* qblk = base + (size_t)t * BLOCKSZ * ld;
    const __bf16* kblk = qblk + HEAD_DIM;
    const __bf16* vblk = qblk + 2 * HEAD_DIM;

    // stage v^T and decayed k^T (coalesced global reads, scattered LDS writes)
    for (int idx = tid; idx < BLOCKSZ * HEAD_DIM; idx += 512) {
      int m = idx >> 7, d = idx & 127;
      vT_s[d][m] = vblk[(size_t)m * ld + d];
      float kv = (float)kblk[(size_t)m * ld + d];
      kdT_s[d][m] = (__bf16)(kv * qexp_s[BLOCKSZ - 1 - m]);   // k_decay(m)
    }
    __syncthreads();

    // q fragments for this wave's 16 rows (reused for inter and scores)
    const __bf16* qrow = qblk + (size_t)(w * 16 + ln) * ld;
    v16bf aq[4];
#pragma unroll
    for (int kk = 0; kk < 4; ++kk) aq[kk] = frag_row_ld(qrow, kk * 32, half);

    // ---- single accumulator: acc = q_decay * (q @ kv) first, intra on top ---
    v8f acc[8];
#pragma unroll
    for (int e = 0; e < 8; ++e) acc[e] = zero8;
#pragma unroll
    for (int e = 0; e < 8; ++e)
#pragma unroll
      for (int kk = 0; kk < 4; ++kk)
        acc[e] = wmma_bf16(aq[kk], frag_row_ld(&kvT_b[e * 16 + ln][0], kk * 32, half), acc[e]);
#pragma unroll
    for (int r = 0; r < 8; ++r) {
      const float qd = qexp_s[w * 16 + half * 8 + r + 1];     // exp(-slope*(i+1))
#pragma unroll
      for (int e = 0; e < 8; ++e) acc[e][r] *= qd;
    }

    // ---- intra-block: causal decayed attention, 32-key chunks ----
    for (int j = 0; j <= (w >> 1); ++j) {                     // causal chunk skip
#pragma unroll
      for (int jt = 0; jt < 2; ++jt) {
        const __bf16* krow = kblk + (size_t)(j * 32 + jt * 16 + ln) * ld;  // B col = k row
        v16bf bkf[4];
#pragma unroll
        for (int kk = 0; kk < 4; ++kk) bkf[kk] = frag_row_ld(krow, kk * 32, half);
        v8f s = zero8;
#pragma unroll
        for (int kk = 0; kk < 4; ++kk) s = wmma_bf16(aq[kk], bkf[kk], s);
        const int jcol = j * 32 + jt * 16 + ln;
#pragma unroll
        for (int r = 0; r < 8; ++r) {                         // diag_decay in C-layout
          int i = w * 16 + half * 8 + r;
          int diff = i - jcol;
          float val = (diff >= 0) ? s[r] * qexp_s[diff & 255] : 0.f;
          sc_s[w][half * 8 + r][jt * 16 + ln] = (__bf16)val;
        }
      }
      // scores (D-layout) -> A-layout via wave-private LDS, then scores @ v
      v16bf as = frag_row_ld(&sc_s[w][ln][0], 0, half);
#pragma unroll
      for (int e = 0; e < 8; ++e) {
        v16bf bv = frag_row_ld(&vT_s[e * 16 + ln][0], j * 32, half);
        acc[e] = wmma_bf16(as, bv, acc[e]);
      }
    }

    // ---- write (b, s, h*d) fp32 ----
    float* obase = attn_out + ((size_t)b * SEQLEN + (size_t)t * BLOCKSZ) * HIDDEN + (size_t)h * HEAD_DIM;
#pragma unroll
    for (int e = 0; e < 8; ++e)
#pragma unroll
      for (int r = 0; r < 8; ++r) {
        int i = w * 16 + half * 8 + r;
        obase[(size_t)i * HIDDEN + e * 16 + ln] = acc[e][r];
      }
    __syncthreads();   // all kv reads complete before update writes

    // ---- kv update: kv^T = kv^T*block_decay + v^T @ (k*k_decay) ----
    // 64 tiles (et, dt); wave owns et = w&7, dt in [4*(w>>3), +4)
    const int et = w & 7, dtb = (w >> 3) * 4;
    v8f c[4];
#pragma unroll
    for (int dd = 0; dd < 4; ++dd)
#pragma unroll
      for (int r = 0; r < 8; ++r)
        c[dd][r] = kvT_f[et * 16 + half * 8 + r][(dtb + dd) * 16 + ln] * block_decay;
#pragma unroll
    for (int kk = 0; kk < 8; ++kk) {   // K = 256 block rows
      v16bf av = frag_row_ld(&vT_s[et * 16 + ln][0], kk * 32, half);
#pragma unroll
      for (int dd = 0; dd < 4; ++dd) {
        v16bf bk = frag_row_ld(&kdT_s[(dtb + dd) * 16 + ln][0], kk * 32, half);
        c[dd] = wmma_bf16(av, bk, c[dd]);
      }
    }
#pragma unroll
    for (int dd = 0; dd < 4; ++dd)
#pragma unroll
      for (int r = 0; r < 8; ++r) {
        kvT_f[et * 16 + half * 8 + r][(dtb + dd) * 16 + ln] = c[dd][r];
        kvT_b[et * 16 + half * 8 + r][(dtb + dd) * 16 + ln] = (__bf16)c[dd][r];
      }
    __syncthreads();   // update done before next block reuses vT/kdT/kv
  }
}

// ---------------- RMSNorm * gate (per token) ----------------
__global__ __launch_bounds__(256) void norm_gate_k(const float* __restrict__ attn,
                                                   const __bf16* __restrict__ gate,
                                                   const float* __restrict__ norm_w,
                                                   __bf16* __restrict__ out) {
  __shared__ float red[8];
  const int token = blockIdx.x;
  const float* x = attn + (size_t)token * HIDDEN;
  float ss = 0.f;
  for (int c = threadIdx.x; c < HIDDEN; c += 256) { float v = x[c]; ss += v * v; }
#pragma unroll
  for (int off = 16; off > 0; off >>= 1) ss += __shfl_down(ss, off, 32);
  if ((threadIdx.x & 31) == 0) red[threadIdx.x >> 5] = ss;
  __syncthreads();
  float tot = 0.f;
#pragma unroll
  for (int i = 0; i < 8; ++i) tot += red[i];
  float rs = rsqrtf(tot / (float)HIDDEN + 1e-5f);
  const __bf16* g = gate + (size_t)token * HIDDEN;
  __bf16* o = out + (size_t)token * HIDDEN;
  for (int c = threadIdx.x; c < HIDDEN; c += 256)
    o[c] = (__bf16)((float)g[c] * norm_w[c] * x[c] * rs);
}

// ---------------- launch ----------------
extern "C" void kernel_launch(void* const* d_in, const int* in_sizes, int n_in,
                              void* d_out, int out_size, void* d_ws, size_t ws_size,
                              hipStream_t stream) {
  const float* hidden = (const float*)d_in[0];   // (2,4096,4096)
  const float* qkv_w  = (const float*)d_in[1];   // (12288,4096)
  const float* out_w  = (const float*)d_in[2];   // (4096,4096)
  const float* gate_w = (const float*)d_in[3];   // (4096,4096)
  const float* norm_w = (const float*)d_in[4];   // (4096)
  float* out = (float*)d_out;

  char* ws = (char*)d_ws;
  size_t off = 0;
  auto take = [&](size_t bytes) { char* p = ws + off; off = (off + bytes + 255) & ~(size_t)255; return p; };
  __bf16* hidden_bf = (__bf16*)take((size_t)TOKENS * HIDDEN * 2);
  __bf16* qkvw_bf   = (__bf16*)take((size_t)QKV_LD * HIDDEN * 2);
  __bf16* gatew_bf  = (__bf16*)take((size_t)HIDDEN * HIDDEN * 2);
  __bf16* outw_bf   = (__bf16*)take((size_t)HIDDEN * HIDDEN * 2);
  __bf16* qkv_bf    = (__bf16*)take((size_t)TOKENS * QKV_LD * 2);
  __bf16* gate_bf   = (__bf16*)take((size_t)TOKENS * HIDDEN * 2);
  float*  attn_f    = (float*) take((size_t)TOKENS * HIDDEN * 4);
  __bf16* normed_bf = (__bf16*)take((size_t)TOKENS * HIDDEN * 2);
  (void)ws_size; (void)in_sizes; (void)n_in; (void)out_size;

  // 1) one-time fp32 -> bf16 conversion (HBM-bound; done once)
  f32_to_bf16_k<<<4096, 256, 0, stream>>>(hidden, hidden_bf, (size_t)TOKENS * HIDDEN);
  f32_to_bf16_k<<<4096, 256, 0, stream>>>(qkv_w,  qkvw_bf,   (size_t)QKV_LD * HIDDEN);
  f32_to_bf16_k<<<4096, 256, 0, stream>>>(gate_w, gatew_bf,  (size_t)HIDDEN * HIDDEN);
  f32_to_bf16_k<<<4096, 256, 0, stream>>>(out_w,  outw_bf,   (size_t)HIDDEN * HIDDEN);

  // 2) qkv = silu(hidden @ qkv_w^T)   [M=8192, N=12288, K=4096]
  gemm_bf16_k<0, __bf16><<<dim3(QKV_LD / 128, TOKENS / 128), 256, 0, stream>>>(
      hidden_bf, qkvw_bf, qkv_bf, TOKENS, QKV_LD, HIDDEN);

  // 3) gate = sigmoid(hidden @ gate_w^T)   [N=4096]
  gemm_bf16_k<1, __bf16><<<dim3(HIDDEN / 128, TOKENS / 128), 256, 0, stream>>>(
      hidden_bf, gatew_bf, gate_bf, TOKENS, HIDDEN, HIDDEN);

  // 4) lightning attention scan: one WG per (b, h)
  lightning_attn_k<<<BATCH * HEADS, 512, 0, stream>>>(qkv_bf, attn_f);

  // 5) RMSNorm * gate
  norm_gate_k<<<TOKENS, 256, 0, stream>>>(attn_f, gate_bf, norm_w, normed_bf);

  // 6) out = normed @ out_w^T -> fp32
  gemm_bf16_k<2, float><<<dim3(HIDDEN / 128, TOKENS / 128), 256, 0, stream>>>(
      normed_bf, outw_bf, out, TOKENS, HIDDEN, HIDDEN);
}
